// MPNN_47124381172062
// MI455X (gfx1250) — matrix operand order repaired and checked by reading, not compiled, not run on hardware
//
#include <hip/hip_runtime.h>
#include <hip/hip_bf16.h>

typedef __attribute__((ext_vector_type(16))) _Float16 v16h;
typedef __attribute__((ext_vector_type(8)))  float    v8f;
typedef __attribute__((ext_vector_type(2)))  _Float16 h2;

#define N_NODES 100000
#define N_EDGES 1600000
#define FEATS   512
#define HIDDEN  64
#define CLASSES 64
#define DEPTHP  10
#define N_RB    (N_NODES / 16)   /* 6250 row-blocks of 16 rows */

// ---------------------------------------------------------------- utility
__global__ void k_zero2(int* __restrict__ a, int* __restrict__ b, int n) {
    int i = blockIdx.x * blockDim.x + threadIdx.x;
    if (i < n) { a[i] = 0; b[i] = 0; }
}

// degree count over dst
__global__ void k_deg(const long long* __restrict__ dst, int* __restrict__ cnt, int E) {
    int e = blockIdx.x * blockDim.x + threadIdx.x;
    if (e < E) atomicAdd(&cnt[(int)dst[e]], 1);
}

__global__ void k_isd(const int* __restrict__ cnt, float* __restrict__ isd, int n) {
    int i = blockIdx.x * blockDim.x + threadIdx.x;
    if (i < n) {
        float d = (float)cnt[i];
        isd[i] = rsqrtf(d > 1.0f ? d : 1.0f);
    }
}

// single-workgroup exclusive scan of cnt[0..N) -> rowptr[0..N], rowptr[N]=E
__global__ void __launch_bounds__(1024) k_scan(const int* __restrict__ cnt,
                                               int* __restrict__ rowptr) {
    __shared__ int sums[1024];
    const int CH = (N_NODES + 1023) / 1024;   // 98
    int t = threadIdx.x;
    int base = t * CH;
    int s = 0;
    for (int i = 0; i < CH; ++i) {
        int g = base + i;
        if (g < N_NODES) s += cnt[g];
    }
    sums[t] = s;
    __syncthreads();
    for (int off = 1; off < 1024; off <<= 1) {
        int u = (t >= off) ? sums[t - off] : 0;
        __syncthreads();
        sums[t] += u;
        __syncthreads();
    }
    int run = sums[t] - s;                    // exclusive prefix of this chunk
    for (int i = 0; i < CH; ++i) {
        int g = base + i;
        if (g < N_NODES) { rowptr[g] = run; run += cnt[g]; }
    }
    if (t == 0) rowptr[N_NODES] = N_EDGES;
}

// CSR fill; per-edge record packs {src, norm} so the hot loop does one
// uniform 8B (s_load_b64) fetch per edge.
__global__ void k_fill(const long long* __restrict__ src, const long long* __restrict__ dst,
                       const int* __restrict__ rowptr, int* __restrict__ cnt2,
                       const float* __restrict__ isd,
                       int2* __restrict__ epair, int E) {
    int e = blockIdx.x * blockDim.x + threadIdx.x;
    if (e >= E) return;
    int s = (int)src[e];
    int d = (int)dst[e];
    int pos = rowptr[d] + atomicAdd(&cnt2[d], 1);
    epair[pos] = make_int2(s, __float_as_int(isd[s] * isd[d]));
}

// ---------------------------------------------------------------- weight pre-pack
// Pack W1/W2 into the exact v_wmma 16x16x32 f16 B-matrix lane layout so that a
// B-fragment is a single contiguous 32B load per (kstep, ntile, lane).
// B 32x16: lanes 0-15 -> N=lane, K=0..7,16..23 ; lanes 16-31 -> N=lane-16, K=8..15,24..31
__global__ void k_wfrag(const float* __restrict__ W1, const float* __restrict__ W2,
                        _Float16* __restrict__ w1f, _Float16* __restrict__ w2f) {
    int idx = blockIdx.x * blockDim.x + threadIdx.x;
    if (idx < 16 * 4 * 32 * 16) {             // FEATS/32 ksteps * 4 ntiles
        int j = idx & 15, lane = (idx >> 4) & 31, n = (idx >> 9) & 3, ks = idx >> 11;
        int k   = ks * 32 + ((lane < 16) ? 0 : 8) + (j & 7) + ((j >> 3) << 4);
        int col = n * 16 + (lane & 15);
        w1f[idx] = (_Float16)W1[k * HIDDEN + col];
    }
    if (idx < 2 * 4 * 32 * 16) {              // HIDDEN/32 ksteps * 4 ntiles
        int j = idx & 15, lane = (idx >> 4) & 31, n = (idx >> 9) & 3, ks = idx >> 11;
        int k   = ks * 32 + ((lane < 16) ? 0 : 8) + (j & 7) + ((j >> 3) << 4);
        int col = n * 16 + (lane & 15);
        w2f[idx] = (_Float16)W2[k * CLASSES + col];
    }
}

// ---------------------------------------------------------------- fused MLP (WMMA)
// One wave32 per 16-row block. GEMM1 (512->64) in f16 WMMA w/ f32 accum,
// bias+relu, through LDS, GEMM2 (64->64), bias, store h0 (f32) and h (f16).
__global__ void __launch_bounds__(256) k_mlp(const float* __restrict__ X,
                                             const _Float16* __restrict__ w1f,
                                             const _Float16* __restrict__ w2f,
                                             const float* __restrict__ b1,
                                             const float* __restrict__ b2,
                                             float* __restrict__ h0,
                                             _Float16* __restrict__ h16) {
    __shared__ float lds[8][16][65];          // per-wave 16x64 tile, padded
    int lane = threadIdx.x & 31;
    int wid  = threadIdx.x >> 5;
    int rb   = blockIdx.x * 8 + wid;
    bool active = rb < N_RB;
    int rbc  = active ? rb : 0;

    int m    = lane & 15;                     // A-frag row / output N index
    int koff = (lane < 16) ? 0 : 8;           // A-frag K sub-block

    // ---- GEMM1: acc[n] = X[16x512] @ W1[:, n*16:(n+1)*16]
    v8f acc[4] = {v8f{0}, v8f{0}, v8f{0}, v8f{0}};
    const float* xrow = X + (size_t)(rbc * 16 + m) * FEATS;
    for (int ks = 0; ks < FEATS / 32; ++ks) {
        int k0 = ks * 32;
        v16h a;
#pragma unroll
        for (int i = 0; i < 8; ++i) {
            a[i]     = (_Float16)xrow[k0 + koff + i];        // K = koff..koff+7
            a[8 + i] = (_Float16)xrow[k0 + 16 + koff + i];   // K = 16+koff..
        }
#pragma unroll
        for (int n = 0; n < 4; ++n) {
            v16h b = *(const v16h*)(w1f + (size_t)((ks * 4 + n) * 32 + lane) * 16);
            acc[n] = __builtin_amdgcn_wmma_f32_16x16x32_f16(
                false, a, false, b, (short)0, acc[n], false, false);
        }
    }

    // ---- bias + relu -> LDS tile (C layout: VGPR r holds rows r / r+8)
#pragma unroll
    for (int n = 0; n < 4; ++n) {
        float bias = b1[n * 16 + m];
#pragma unroll
        for (int r = 0; r < 8; ++r) {
            float v = acc[n][r] + bias;
            v = v > 0.0f ? v : 0.0f;
            int row = (lane < 16) ? r : r + 8;
            lds[wid][row][n * 16 + m] = v;
        }
    }
    __syncthreads();

    // ---- GEMM2: out = relu_tile[16x64] @ W2
    v8f acc2[4] = {v8f{0}, v8f{0}, v8f{0}, v8f{0}};
#pragma unroll
    for (int ks = 0; ks < HIDDEN / 32; ++ks) {
        int k0 = ks * 32;
        v16h a;
#pragma unroll
        for (int i = 0; i < 8; ++i) {
            a[i]     = (_Float16)lds[wid][m][k0 + koff + i];
            a[8 + i] = (_Float16)lds[wid][m][k0 + 16 + koff + i];
        }
#pragma unroll
        for (int n = 0; n < 4; ++n) {
            v16h b = *(const v16h*)(w2f + (size_t)((ks * 4 + n) * 32 + lane) * 16);
            acc2[n] = __builtin_amdgcn_wmma_f32_16x16x32_f16(
                false, a, false, b, (short)0, acc2[n], false, false);
        }
    }

    if (active) {
        float*     out  = h0  + (size_t)rb * 16 * CLASSES;
        _Float16*  out2 = h16 + (size_t)rb * 16 * CLASSES;
#pragma unroll
        for (int n = 0; n < 4; ++n) {
            float bias = b2[n * 16 + m];
#pragma unroll
            for (int r = 0; r < 8; ++r) {
                int row = (lane < 16) ? r : r + 8;
                float v = acc2[n][r] + bias;
                out [row * CLASSES + n * 16 + m] = v;
                out2[row * CLASSES + n * 16 + m] = (_Float16)v;
            }
        }
    }
}

// ---------------------------------------------------------------- propagation
// One wave32 per node; lane holds 2 feats. Gather-sum over CSR, no atomics.
// Working h is f16 (halves the L2 gather traffic: 128B/edge instead of 256B),
// accumulation + alpha-mix in f32; final round writes f32 into d_out.
__global__ void __launch_bounds__(256) k_prop(const _Float16* __restrict__ hin,
                                              const float* __restrict__ h0,
                                              const int* __restrict__ rowptr,
                                              const int2* __restrict__ epair,
                                              _Float16* __restrict__ out16,
                                              float* __restrict__ out32) {
    int lane = threadIdx.x & 31;
    int node = blockIdx.x * 8 + (threadIdx.x >> 5);
    if (node >= N_NODES) return;
    int beg = rowptr[node];
    int end = rowptr[node + 1];
    float ax = 0.0f, ay = 0.0f;
    for (int j = beg; j < end; ++j) {
        int2  e = epair[j];                   // uniform per wave -> s_load_b64
        int   s = e.x;
        float w = __int_as_float(e.y);
        h2 v = ((const h2*)(hin + (size_t)s * CLASSES))[lane];
        ax += w * (float)v.x;
        ay += w * (float)v.y;
    }
    float2 z = ((const float2*)(h0 + (size_t)node * CLASSES))[lane];
    float ox = 0.9f * ax + 0.1f * z.x;
    float oy = 0.9f * ay + 0.1f * z.y;
    if (out32) {
        float2 o; o.x = ox; o.y = oy;
        ((float2*)(out32 + (size_t)node * CLASSES))[lane] = o;
    } else {
        h2 o; o.x = (_Float16)ox; o.y = (_Float16)oy;
        ((h2*)(out16 + (size_t)node * CLASSES))[lane] = o;
    }
}

// ---------------------------------------------------------------- launch
extern "C" void kernel_launch(void* const* d_in, const int* in_sizes, int n_in,
                              void* d_out, int out_size, void* d_ws, size_t ws_size,
                              hipStream_t stream) {
    const float*     x     = (const float*)d_in[0];
    const long long* edges = (const long long*)d_in[1];   // int64 per reference
    const float*     W1    = (const float*)d_in[2];
    const float*     b1    = (const float*)d_in[3];
    const float*     W2    = (const float*)d_in[4];
    const float*     b2    = (const float*)d_in[5];
    const long long* esrc64 = edges;
    const long long* edst64 = edges + N_EDGES;

    char* w = (char*)d_ws;
    size_t off = 0;
    auto alloc = [&](size_t bytes) -> void* {
        void* p = (void*)(w + off);
        off = (off + bytes + 255) & ~(size_t)255;
        return p;
    };
    int*      cnt    = (int*)alloc(sizeof(int) * N_NODES);
    int*      cnt2   = (int*)alloc(sizeof(int) * N_NODES);
    int*      rowptr = (int*)alloc(sizeof(int) * (N_NODES + 1));
    float*    isd    = (float*)alloc(sizeof(float) * N_NODES);
    int2*     epair  = (int2*)alloc(sizeof(int2) * N_EDGES);
    _Float16* w1f    = (_Float16*)alloc(sizeof(_Float16) * 16 * 4 * 32 * 16);
    _Float16* w2f    = (_Float16*)alloc(sizeof(_Float16) * 2 * 4 * 32 * 16);
    float*    h0buf  = (float*)alloc(sizeof(float) * (size_t)N_NODES * CLASSES);
    _Float16* hA16   = (_Float16*)alloc(sizeof(_Float16) * (size_t)N_NODES * CLASSES);
    _Float16* hB16   = (_Float16*)alloc(sizeof(_Float16) * (size_t)N_NODES * CLASSES);

    const int TB = 256;
    int gN = (N_NODES + TB - 1) / TB;
    int gE = (N_EDGES + TB - 1) / TB;

    k_zero2<<<gN, TB, 0, stream>>>(cnt, cnt2, N_NODES);
    k_deg  <<<gE, TB, 0, stream>>>(edst64, cnt, N_EDGES);
    k_isd  <<<gN, TB, 0, stream>>>(cnt, isd, N_NODES);
    k_scan <<<1, 1024, 0, stream>>>(cnt, rowptr);
    k_fill <<<gE, TB, 0, stream>>>(esrc64, edst64, rowptr, cnt2, isd, epair, N_EDGES);
    k_wfrag<<<(16 * 4 * 32 * 16 + TB - 1) / TB, TB, 0, stream>>>(W1, W2, w1f, w2f);
    k_mlp  <<<(N_RB + 7) / 8, TB, 0, stream>>>(x, w1f, w2f, b1, b2, h0buf, hA16);

    // rounds 1..9 ping-pong f16 buffers; round 10 writes f32 into d_out
    const _Float16* cur = hA16;
    int gP = (N_NODES + 7) / 8;
    for (int it = 1; it <= DEPTHP; ++it) {
        if (it == DEPTHP) {
            k_prop<<<gP, TB, 0, stream>>>(cur, h0buf, rowptr, epair,
                                          (_Float16*)nullptr, (float*)d_out);
        } else {
            _Float16* o = (it & 1) ? hB16 : hA16;
            k_prop<<<gP, TB, 0, stream>>>(cur, h0buf, rowptr, epair, o, (float*)nullptr);
            cur = o;
        }
    }
}
